// MambaBlock_36301063586179
// MI455X (gfx1250) — matrix operand order repaired and checked by reading, not compiled, not run on hardware
//
#include <hip/hip_runtime.h>
#include <hip/hip_bf16.h>
#include <math.h>

// ---------------------------------------------------------------------------
// MambaBlock for MI455X (gfx1250, wave32, WMMA).
// B=2, L=2048, D_IN=512, D_MODEL=1024, D_STATE=16, D_DISCR=64, K=4
// GEMMs: v_wmma_f32_16x16x32_bf16, 4 N-tiles per wave (A-fragment reuse),
// double-buffered A-tile staged to LDS via GLOBAL_LOAD_ASYNC_TO_LDS_B32,
// k-loop unrolled x2 so buffer parity is compile-time (no pointer selects),
// branch-free inner tile loop (clamped B row instead of predication).
// Sequential selective scan: one lane per (channel,state).
// ---------------------------------------------------------------------------

typedef __bf16 v16bf __attribute__((ext_vector_type(16)));
typedef float  v8f   __attribute__((ext_vector_type(8)));

#if __has_builtin(__builtin_amdgcn_global_load_async_to_lds_b32) && \
    __has_builtin(__builtin_amdgcn_s_wait_asynccnt)
#define USE_ASYNC_LDS 1
#else
#define USE_ASYNC_LDS 0
#endif

template <int EPI>
__device__ inline float gemm_epilogue(float x, const float* bias, int n) {
  if (EPI == 1) {               // softplus(x + bias[n])
    x += bias[n];
    x = (x > 20.f) ? x : log1pf(__expf(x));
  }
  return x;
}

__device__ inline void pack4(v16bf& f, int base, float4 q) {
  f[base + 0] = (__bf16)q.x;
  f[base + 1] = (__bf16)q.y;
  f[base + 2] = (__bf16)q.z;
  f[base + 3] = (__bf16)q.w;
}

// out[m,n] = epi( sum_k A[m,k] * W[n,k] )
// A: M x K f32 (stride K).  W: N x K f32 (stride K).  out: M x N f32.
// M % 16 == 0, N % 16 == 0, K % 64 == 0.
// Block = 256 threads = 8 waves; each wave owns 4 16x16 tiles (16 x 64 of out);
// block covers rows [m0, m0+16) x cols [blockIdx.x*512, +512).
template <int EPI>
__global__ void __launch_bounds__(256)
wmma_gemm(const float* __restrict__ A, const float* __restrict__ W,
          const float* __restrict__ bias, float* __restrict__ out,
          int M, int N, int K) {
  constexpr int NT = 4;             // N tiles per wave
  const int tid    = threadIdx.x;
  const int wave   = tid >> 5;
  const int lane   = tid & 31;
  const int lrow   = lane & 15;     // row within fragment half
  const int hi     = lane >> 4;     // which K-half this lane carries
  const int m0     = blockIdx.y * 16;
  const int waveN0 = (blockIdx.x * 8 + wave) * (16 * NT);

  // Wave-uniform count of valid 16-wide tiles (used only in the epilogue).
  int nact = (N - waveN0 + 15) >> 4;
  nact = nact < 0 ? 0 : (nact > NT ? NT : nact);
  nact = __builtin_amdgcn_readfirstlane(nact);

  // Double-buffered A tile, raw f32: 16 rows x 32 k, row stride 36 (144 B).
  __shared__ __attribute__((aligned(16))) float Asf[2][16 * 36];
  float* const b0 = &Asf[0][0];
  float* const b1 = &Asf[1][0];

  v8f acc[NT] = {};

  // Staging assignment: thread stages elements (sr,sc) and (sr+8,sc).
  const int sr = tid >> 5;          // 0..7
  const int sc = tid & 31;          // 0..31

  auto stage = [&](int k0, float* buf) {
    const float* g0 = &A[(size_t)(m0 + sr)     * K + k0 + sc];
    const float* g1 = &A[(size_t)(m0 + sr + 8) * K + k0 + sc];
#if USE_ASYNC_LDS
    __builtin_amdgcn_global_load_async_to_lds_b32(
        (int*)g0, (int*)&buf[sr * 36 + sc], 0, 0);
    __builtin_amdgcn_global_load_async_to_lds_b32(
        (int*)g1, (int*)&buf[(sr + 8) * 36 + sc], 0, 0);
#else
    buf[sr * 36 + sc]       = *g0;
    buf[(sr + 8) * 36 + sc] = *g1;
#endif
  };

  // Branch-free 32-deep k-step on a fixed LDS buffer: build the A fragment,
  // then 4 unconditional WMMAs with the B row clamped (lane-level min).
  auto compute = [&](int k0, const float* buf) {
    // A fragment per documented 16-bit 16x32 layout:
    //   lanes 0-15 : row=lane,    K = {0..7, 16..23}
    //   lanes 16-31: row=lane-16, K = {8..15, 24..31}
    const int kk = hi ? 8 : 0;
    v16bf afrag;
    pack4(afrag, 0,  *(const float4*)&buf[lrow * 36 + kk]);
    pack4(afrag, 4,  *(const float4*)&buf[lrow * 36 + kk + 4]);
    pack4(afrag, 8,  *(const float4*)&buf[lrow * 36 + kk + 16]);
    pack4(afrag, 12, *(const float4*)&buf[lrow * 36 + kk + 20]);
    #pragma unroll
    for (int t = 0; t < NT; ++t) {
      // B (KxN) from W stored N x K: lane n -> column; row clamped so tail
      // waves do harmless redundant math instead of branching.
      int wr = waveN0 + t * 16 + lrow;
      wr = wr < N ? wr : N - 1;
      const float4* wp4 =
          (const float4*)(W + (size_t)wr * K + k0 + (hi ? 8 : 0));
      v16bf bfrag;
      pack4(bfrag, 0,  wp4[0]);
      pack4(bfrag, 4,  wp4[1]);
      pack4(bfrag, 8,  wp4[4]);
      pack4(bfrag, 12, wp4[5]);
      acc[t] = __builtin_amdgcn_wmma_f32_16x16x32_bf16(
          /*neg_a=*/false, afrag, /*neg_b=*/false, bfrag,
          /*c_mod=*/(short)0, acc[t], /*reuse_a=*/false, /*reuse_b=*/false);
    }
  };

  stage(0, b0);                     // prologue: DMA first tile into buffer 0

  for (int k0 = 0; k0 < K; k0 += 64) {
    // ---- first half: compute b0, DMA k0+32 into b1 (K%64==0 => in range)
    stage(k0 + 32, b1);
#if USE_ASYNC_LDS
    // Async loads complete in order: <=2 outstanding means b0's pair landed.
    __builtin_amdgcn_s_wait_asynccnt(2);
#endif
    __syncthreads();
    compute(k0, b0);
    __syncthreads();

    // ---- second half: compute b1, DMA k0+64 into b0 if there is more K
    const bool more = (k0 + 64) < K;
    if (more) stage(k0 + 64, b0);
#if USE_ASYNC_LDS
    if (more) __builtin_amdgcn_s_wait_asynccnt(2);
    else      __builtin_amdgcn_s_wait_asynccnt(0);
#endif
    __syncthreads();
    compute(k0 + 32, b1);
    __syncthreads();
  }

  // C/D layout: VGPR r -> row m0 + r (+8 for lanes 16-31), col n0 + (lane&15).
  #pragma unroll
  for (int t = 0; t < NT; ++t) {
    if (t < nact) {                 // uniform SGPR compare -> s_cbranch
      const int col = waveN0 + t * 16 + lrow;
      #pragma unroll
      for (int r = 0; r < 8; ++r) {
        const int row = m0 + r + (hi ? 8 : 0);
        out[(size_t)row * N + col] = gemm_epilogue<EPI>(acc[t][r], bias, col);
      }
    }
  }
}

// Depthwise causal conv (K=4, left pad 3) + bias + SiLU.
// ab: (BL, 2048) -- a-branch is cols [0,1024). out: (BL, 1024).
__global__ void __launch_bounds__(256)
conv_silu_kernel(const float* __restrict__ ab, const float* __restrict__ cw,
                 const float* __restrict__ cb, float* __restrict__ out,
                 int BL, int L) {
  const int idx = blockIdx.x * 256 + threadIdx.x;   // over BL*1024
  if (idx >= BL * 1024) return;
  const int d  = idx & 1023;
  const int bl = idx >> 10;
  const int l  = bl % L;
  float acc = cb[d];
  #pragma unroll
  for (int j = 0; j < 4; ++j) {
    const int lp = l - 3 + j;
    if (lp >= 0) acc += cw[d * 4 + j] * ab[(size_t)(bl - (3 - j)) * 2048 + d];
  }
  const float s = acc / (1.f + __expf(-acc));       // SiLU
  out[(size_t)bl * 1024 + d] = s;
}

// Selective scan: one lane per (channel, state); 2 channels per wave.
// h_t = exp(delta*A)*h_{t-1} + delta*B*a ; y = sum_s h*C ; fused skip + gate.
__global__ void __launch_bounds__(256)
scan_kernel(const float* __restrict__ delta, const float* __restrict__ aConv,
            const float* __restrict__ Bm, const float* __restrict__ Cm,
            const float* __restrict__ ab /* gate at col offset 1024 */,
            const float* __restrict__ A_log, const float* __restrict__ D_skip,
            float* __restrict__ z, int L) {
  const int tid  = threadIdx.x;
  const int lane = tid & 31;
  const int wave = tid >> 5;
  const int s    = lane & 15;           // state index
  const int half = lane >> 4;           // which channel of the wave's pair
  const int channel = blockIdx.x * 16 + wave * 2 + half;  // 0..2047
  const int b = channel >> 10;
  const int d = channel & 1023;

  const float Aneg = -__expf(A_log[d * 16 + s]);
  const float dsk  = D_skip[d];
  const size_t chanBase = (size_t)b * L;

  float h = 0.f;
  for (int t = 0; t < L; ++t) {
    const size_t row = chanBase + t;
    const float dv = delta[row * 1024 + d];
    const float av = aConv[row * 1024 + d];
    const float Bv = Bm[row * 16 + s];
    const float Cv = Cm[row * 16 + s];
    if (t + 16 < L) {                   // speculative prefetch (global_prefetch_b8)
      __builtin_prefetch(&delta[(row + 16) * 1024 + d], 0, 0);
      __builtin_prefetch(&aConv[(row + 16) * 1024 + d], 0, 0);
    }
    h = __expf(dv * Aneg) * h + (dv * Bv) * av;
    float p = h * Cv;                   // reduce 16 states (stays inside group)
    p += __shfl_xor(p, 1);
    p += __shfl_xor(p, 2);
    p += __shfl_xor(p, 4);
    p += __shfl_xor(p, 8);
    if (s == 0) {
      const float g  = ab[row * 2048 + 1024 + d];
      const float yv = p + dsk * av;
      z[row * 1024 + d] = yv * (g / (1.f + __expf(-g)));
    }
  }
}

extern "C" void kernel_launch(void* const* d_in, const int* in_sizes, int n_in,
                              void* d_out, int out_size, void* d_ws, size_t ws_size,
                              hipStream_t stream) {
  (void)in_sizes; (void)n_in; (void)out_size; (void)ws_size;
  const float* seq    = (const float*)d_in[0];
  const float* W_in   = (const float*)d_in[1];
  const float* W_out  = (const float*)d_in[2];
  const float* W_B    = (const float*)d_in[3];
  const float* W_C    = (const float*)d_in[4];
  const float* W_D1   = (const float*)d_in[5];
  const float* W_D2   = (const float*)d_in[6];
  const float* b_D2   = (const float*)d_in[7];
  const float* conv_w = (const float*)d_in[8];
  const float* conv_b = (const float*)d_in[9];
  const float* A_log  = (const float*)d_in[10];
  const float* D_skip = (const float*)d_in[11];
  float* out = (float*)d_out;

  constexpr int Bb = 2, L = 2048, DIN = 512, DM = 1024, DS = 16, DD = 64;
  constexpr int M = Bb * L;  // 4096 token rows

  // Workspace layout (floats): ~85.5 MB total.
  float* ws    = (float*)d_ws;
  float* ab    = ws;                            // M * 2048
  float* aConv = ab    + (size_t)M * 2 * DM;    // M * 1024
  float* x1    = aConv + (size_t)M * DM;        // M * 64
  float* delta = x1    + (size_t)M * DD;        // M * 1024
  float* Bm    = delta + (size_t)M * DM;        // M * 16
  float* Cm    = Bm    + (size_t)M * DS;        // M * 16
  float* z     = Cm    + (size_t)M * DS;        // M * 1024

  const dim3 blk(256);
  auto gx = [](int N) { return (N + 511) / 512; };  // block covers 512 cols

  // 1) ab = seq @ W_in^T                         (4096 x 2048 x 512)
  wmma_gemm<0><<<dim3(gx(2 * DM), M / 16), blk, 0, stream>>>(
      seq, W_in, nullptr, ab, M, 2 * DM, DIN);
  // 2) a = SiLU(causal depthwise conv(a-branch))
  conv_silu_kernel<<<(M * DM) / 256, blk, 0, stream>>>(ab, conv_w, conv_b, aConv, M, L);
  // 3) x1 = a @ W_D1^T                           (4096 x 64 x 1024)
  wmma_gemm<0><<<dim3(gx(DD), M / 16), blk, 0, stream>>>(aConv, W_D1, nullptr, x1, M, DD, DM);
  // 4) delta = softplus(x1 @ W_D2^T + b_D2)      (4096 x 1024 x 64)
  wmma_gemm<1><<<dim3(gx(DM), M / 16), blk, 0, stream>>>(x1, W_D2, b_D2, delta, M, DM, DD);
  // 5) Bm = a @ W_B^T ; Cm = a @ W_C^T           (4096 x 16 x 1024)
  wmma_gemm<0><<<dim3(gx(DS), M / 16), blk, 0, stream>>>(aConv, W_B, nullptr, Bm, M, DS, DM);
  wmma_gemm<0><<<dim3(gx(DS), M / 16), blk, 0, stream>>>(aConv, W_C, nullptr, Cm, M, DS, DM);
  // 6) selective scan + skip + gate -> z
  scan_kernel<<<(Bb * DM) / 16, blk, 0, stream>>>(delta, aConv, Bm, Cm, ab,
                                                  A_log, D_skip, z, L);
  // 7) out = z @ W_out^T                         (4096 x 512 x 1024)
  wmma_gemm<0><<<dim3(gx(DIN), M / 16), blk, 0, stream>>>(z, W_out, nullptr, out, M, DIN, DM);
}
